// FeedForward2_80771154968813
// MI455X (gfx1250) — matrix operand order repaired and checked by reading, not compiled, not run on hardware
//
#include <hip/hip_runtime.h>
#include <hip/hip_bf16.h>

// MI455X / gfx1250: wave32, WMMA 16x16x32 bf16 -> f32 accum,
// async-to-LDS staging (ACT) + Tensor Data Mover staging (BIL).
typedef __bf16 v16bf __attribute__((ext_vector_type(16)));
typedef float  v8f   __attribute__((ext_vector_type(8)));
typedef unsigned int v4u __attribute__((ext_vector_type(4)));
typedef int          v8i __attribute__((ext_vector_type(8)));
typedef int          v4i __attribute__((ext_vector_type(4)));

// Problem constants (B=2, N=2048, W=1024, A=Bd=128)
constexpr int MTOK = 4096;         // B*N tokens
constexpr int WDIM = 1024;         // hidden width / output width
constexpr int ADIM = 128;          // A == Bd
constexpr int KBIL = ADIM * ADIM;  // 16384 flattened (i,j)

// ---------------------------------------------------------------------------
// Kernel 0: fp32 -> bf16 convert + transpose.  in[R,C] f32 -> out[C,R] bf16.
// ---------------------------------------------------------------------------
__global__ void prep_transpose_cvt(const float* __restrict__ in,
                                   __bf16* __restrict__ out,
                                   int R, int C) {
  __shared__ float t[32][33];
  const int c0 = blockIdx.x * 32, r0 = blockIdx.y * 32;
  const int tx = threadIdx.x, ty = threadIdx.y;
  #pragma unroll
  for (int rr = ty; rr < 32; rr += 8)
    t[rr][tx] = in[(size_t)(r0 + rr) * C + (c0 + tx)];
  __syncthreads();
  #pragma unroll
  for (int rr = ty; rr < 32; rr += 8)
    out[(size_t)(c0 + rr) * R + (r0 + tx)] = (__bf16)t[tx][rr];
}

// ---------------------------------------------------------------------------
// Kernel 1: activations.  y = swish(x @ w + bias).
// w-slice (128 rows x 32 k = 8KB bf16) staged once per block via gfx1250
// async-to-LDS copies, double-buffered against the WMMAs.  B-fragment ds
// reads are software-pipelined one deep so DS latency overlaps the WMMA.
// ---------------------------------------------------------------------------
__global__ void act_gemm_swish(const float*  __restrict__ x,
                               const __bf16* __restrict__ w1t,
                               const __bf16* __restrict__ w2t,
                               const float*  __restrict__ b1,
                               const float*  __restrict__ b2,
                               __bf16* __restrict__ aout,
                               __bf16* __restrict__ bout) {
  constexpr int ROWB = 80;                           // padded row bytes (40 bf16)
  __shared__ __attribute__((aligned(32))) unsigned char wts[2][128 * ROWB]; // 20KB

  const __bf16* wt   = blockIdx.y ? w2t : w1t;
  const float*  bias = blockIdx.y ? b2  : b1;
  __bf16*       out  = blockIdx.y ? bout : aout;

  const int tid  = threadIdx.x;
  const int lane = tid & 31;
  const int wid  = tid >> 5;
  const int hi   = lane >> 4;
  const int m0   = blockIdx.x * 128 + wid * 16;
  const int row  = m0 + (lane & 15);

  const unsigned ldsb[2] = {(unsigned)(uintptr_t)(void*)&wts[0][0],
                            (unsigned)(uintptr_t)(void*)&wts[1][0]};

  // stage one 8KB w-slice: 512 x 16B async transfers, 2 per thread
  auto issue_wt = [&](int k0, unsigned base) {
    #pragma unroll
    for (int c = 0; c < 2; ++c) {
      const int g = c * 256 + tid;                   // 4 chunks per 64B row
      const int r = g >> 2, col = g & 3;
      const unsigned lds_addr = base + r * ROWB + col * 16;
      const unsigned long long ga = (unsigned long long)(uintptr_t)(
          wt + (size_t)r * WDIM + k0 + col * 8);
      asm volatile("global_load_async_to_lds_b128 %0, %1, off"
                   :: "v"(lds_addr), "v"(ga) : "memory");
    }
  };

  issue_wt(0, ldsb[0]);

  v8f acc[8] = {};
  const float* xrow = x + (size_t)row * WDIM;

  for (int k0 = 0; k0 < WDIM; k0 += 32) {
    asm volatile("s_wait_asynccnt 0x0" ::: "memory");
    __syncthreads();
    if (k0 + 32 < WDIM) issue_wt(k0 + 32, ldsb[(k0 / 32 + 1) & 1]);

    // A fragment (x row -> bf16); overlaps with the async prefetch above
    v16bf af;
    const int koff = k0 + hi * 16;
    #pragma unroll
    for (int e = 0; e < 8; ++e) af[e]     = (__bf16)xrow[koff + e];
    #pragma unroll
    for (int e = 0; e < 8; ++e) af[8 + e] = (__bf16)xrow[koff + 8 + e];

    const unsigned char* buf = &wts[(k0 / 32) & 1][0];
    auto ldfrag = [&](int nt) {
      v16bf f;
      __builtin_memcpy(&f, buf + (nt * 16 + (lane & 15)) * ROWB + hi * 32, 32);
      return f;
    };

    v16bf frag[2];
    frag[0] = ldfrag(0);
    #pragma unroll
    for (int nt = 0; nt < 8; ++nt) {                 // 1-deep ds pipeline
      if (nt < 7) frag[(nt + 1) & 1] = ldfrag(nt + 1);
      acc[nt] = __builtin_amdgcn_wmma_f32_16x16x32_bf16(
          false, af, false, frag[nt & 1], (short)0, acc[nt], false, false);
    }
  }

  #pragma unroll
  for (int nt = 0; nt < 8; ++nt) {
    const int col = nt * 16 + (lane & 15);
    const float bv = bias[col];
    #pragma unroll
    for (int r = 0; r < 8; ++r) {
      const float c  = acc[nt][r] + bv;
      const float sw = c / (1.0f + __expf(-c));
      out[(size_t)(m0 + r + hi * 8) * ADIM + col] = (__bf16)sw;
    }
  }
}

// ---------------------------------------------------------------------------
// TDM: one descriptor loads a 128(w) x 128(k) bf16 tile of w3t into LDS with
// hardware row padding: pad_interval = 64 dwords (256B row), pad_amount = 4
// dwords (16B) -> 272B padded rows, matching the ds_load_b128 swizzle.
// D# bitfields per CDNA5 ISA 8.3/8.4 (count=1, type=2, data_size=2B).
// This toolchain exposes the 6-arg builtin (g0, g1, g2, g3, g4, cpol).
// ---------------------------------------------------------------------------
__device__ __forceinline__ void tdm_issue_w3(const __bf16* w3t, int wbase,
                                             int i, unsigned lds_addr) {
  const unsigned long long ga = (unsigned long long)(uintptr_t)(
      w3t + ((size_t)wbase * KBIL + (size_t)i * ADIM));
  v4u g0;
  g0[0] = 1u;                                        // count=1 (valid user D#)
  g0[1] = lds_addr;                                  // lds_addr [63:32]
  g0[2] = (unsigned)(ga & 0xffffffffu);              // global_addr[31:0]
  g0[3] = (unsigned)((ga >> 32) & 0x01ffffffu)       // global_addr[56:32]
        | (2u << 30);                                // type = 2 ("image")
  v8i g1;
  g1[0] = (int)((1u << 16)                           // data_size = 2 bytes
              | (1u << 20)                           // pad_enable
              | (5u << 22)                           // pad_interval: 64 dwords
              | (3u << 25));                         // pad_amount: 4 dwords
  g1[1] = (int)(16384u << 16);                       // tensor_dim0 = 16384
  g1[2] = (int)(1024u << 16);                        // tensor_dim1 = 1024
  g1[3] = (int)(128u << 16);                         // tile_dim0 = 128
  g1[4] = 128;                                       // tile_dim1 = 128
  g1[5] = 16384;                                     // tensor_dim0_stride lo32
  g1[6] = 0;                                         // stride hi | dim1_stride
  g1[7] = 0;
  v4i g2 = {0, 0, 0, 0};                             // 2D tile: groups 2/3 off
  v4i g3 = {0, 0, 0, 0};
  v8i g4 = {0, 0, 0, 0, 0, 0, 0, 0};                 // unused trailing group
  __builtin_amdgcn_tensor_load_to_lds(g0, g1, g2, g3, g4, 0);
}

// ---------------------------------------------------------------------------
// Kernel 2: bilinear GEMM  out[m,w] = sum_k (a[m,i]*b[m,j]) * w3t[w,k],
// k = i*128+j.  Per i-step the 32KB w3t slice is DMA'd into LDS by the TDM
// (wave 0 issues, TENSORcnt-tracked), double-buffered against the 16 WMMAs.
// Fragment ds reads are software-pipelined one deep across the (jt,nt) chain.
// ---------------------------------------------------------------------------
__global__ void bilinear_wmma(const __bf16* __restrict__ aact,
                              const __bf16* __restrict__ bact,
                              const __bf16* __restrict__ w3t,
                              const float*  __restrict__ b3,
                              float* __restrict__ out) {
  constexpr int ROWB = 272;                          // padded row bytes
  __shared__ __attribute__((aligned(32))) unsigned char w3s[2][128 * ROWB]; // 68KB
  __shared__ __bf16 a_s[ADIM * 64];                  // [i][row], 16KB

  const int tid  = threadIdx.x;
  const int lane = tid & 31;
  const int wid  = tid >> 5;
  const int mbase = blockIdx.x * 64;
  const int wbase = blockIdx.y * 128;

  const unsigned ldsb[2] = {(unsigned)(uintptr_t)(void*)&w3s[0][0],
                            (unsigned)(uintptr_t)(void*)&w3s[1][0]};

  if (wid == 0) tdm_issue_w3(w3t, wbase, 0, ldsb[0]);   // prefetch slice 0

  // stage a-tile transposed ([i][row]) while the TDM runs
  for (int e = tid; e < 64 * ADIM; e += 256) {
    const int row = e & 63, col = e >> 6;
    a_s[col * 64 + row] = aact[(size_t)(mbase + row) * ADIM + col];
  }

  const int m_sub = wid & 3, w_sub = wid >> 2;
  const int hi = lane >> 4;
  const int rlocal = m_sub * 16 + (lane & 15);
  const int w0 = wbase + w_sub * 64;

  // preload b fragments for 4 j-tiles (A-matrix 16x32 lane layout), as f32
  float bf[4][16];
  const __bf16* brow = bact + (size_t)(mbase + rlocal) * ADIM;
  #pragma unroll
  for (int jt = 0; jt < 4; ++jt) {
    const int koff = jt * 32 + hi * 16;
    #pragma unroll
    for (int e = 0; e < 8; ++e) bf[jt][e]     = (float)brow[koff + e];
    #pragma unroll
    for (int e = 0; e < 8; ++e) bf[jt][8 + e] = (float)brow[koff + 8 + e];
  }

  v8f acc[4] = {};

  for (int i = 0; i < ADIM; ++i) {
    if (wid == 0) __builtin_amdgcn_s_wait_tensorcnt(0);   // slice i in LDS
    __syncthreads();                                      // publish to block
    if (wid == 0 && i + 1 < ADIM)
      tdm_issue_w3(w3t, wbase, i + 1, ldsb[(i + 1) & 1]); // overlap DMA

    const unsigned char* buf = &w3s[i & 1][0];
    const float s = (float)a_s[i * 64 + rlocal];          // a[m,i]

    // flattened (jt,nt) chain, fragment reads pipelined one deep
    auto ldfrag = [&](int u) {
      const int jt = u >> 2, nt = u & 3;
      const int wrow = w_sub * 64 + nt * 16 + (lane & 15);
      v16bf f;
      __builtin_memcpy(&f, buf + wrow * ROWB + (jt * 32 + hi * 16) * 2, 32);
      return f;
    };

    v16bf frag[2];
    frag[0] = ldfrag(0);
    v16bf af;
    #pragma unroll
    for (int u = 0; u < 16; ++u) {
      if (u < 15) frag[(u + 1) & 1] = ldfrag(u + 1);
      if ((u & 3) == 0) {                               // new j-tile: A = s*b
        const int jt = u >> 2;
        #pragma unroll
        for (int e = 0; e < 16; ++e) af[e] = (__bf16)(s * bf[jt][e]);
      }
      acc[u & 3] = __builtin_amdgcn_wmma_f32_16x16x32_bf16(
          false, af, false, frag[u & 1], (short)0, acc[u & 3], false, false);
    }
  }

  #pragma unroll
  for (int nt = 0; nt < 4; ++nt) {
    const int wcol = w0 + nt * 16 + (lane & 15);
    const float bv = b3[wcol];
    #pragma unroll
    for (int r = 0; r < 8; ++r) {
      const int m = mbase + m_sub * 16 + r + hi * 8;
      out[(size_t)m * WDIM + wcol] = acc[nt][r] + bv;
    }
  }
}

// ---------------------------------------------------------------------------
// Host launcher
// ---------------------------------------------------------------------------
extern "C" void kernel_launch(void* const* d_in, const int* in_sizes, int n_in,
                              void* d_out, int out_size, void* d_ws, size_t ws_size,
                              hipStream_t stream) {
  (void)in_sizes; (void)n_in; (void)out_size; (void)ws_size;
  const float* x  = (const float*)d_in[0];
  const float* w1 = (const float*)d_in[1];
  const float* w2 = (const float*)d_in[2];
  const float* w3 = (const float*)d_in[3];
  const float* b1 = (const float*)d_in[4];
  const float* b2 = (const float*)d_in[5];
  const float* b3 = (const float*)d_in[6];
  float* out = (float*)d_out;

  // workspace (bf16): w3t 32MB | w1t 256KB | w2t 256KB | a 1MB | b 1MB
  char* ws = (char*)d_ws;
  __bf16* w3t  = (__bf16*)ws;
  __bf16* w1t  = (__bf16*)(ws + (size_t)KBIL * WDIM * 2);
  __bf16* w2t  = w1t + (size_t)ADIM * WDIM;
  __bf16* aact = w2t + (size_t)ADIM * WDIM;
  __bf16* bact = aact + (size_t)MTOK * ADIM;

  prep_transpose_cvt<<<dim3(ADIM / 32, WDIM / 32), dim3(32, 8), 0, stream>>>(
      w1, w1t, WDIM, ADIM);
  prep_transpose_cvt<<<dim3(ADIM / 32, WDIM / 32), dim3(32, 8), 0, stream>>>(
      w2, w2t, WDIM, ADIM);
  prep_transpose_cvt<<<dim3(WDIM / 32, KBIL / 32), dim3(32, 8), 0, stream>>>(
      w3, w3t, KBIL, WDIM);

  act_gemm_swish<<<dim3(MTOK / 128, 2), 256, 0, stream>>>(
      x, w1t, w2t, b1, b2, aact, bact);

  bilinear_wmma<<<dim3(MTOK / 64, WDIM / 128), 256, 0, stream>>>(
      aact, bact, w3t, b3, out);
}
